// SAE_30717606101583
// MI455X (gfx1250) — matrix operand order repaired and checked by reading, not compiled, not run on hardware
//
#include <hip/hip_runtime.h>
#include <hip/hip_bf16.h>
#include <stdint.h>

#define BATCH 4096
#define DIN   2048
#define NLAT  65536
#define TOPK  64
#define SLICE 512                  // latent slice per block (per-slice top-64)
#define MT    128                  // batch rows per block
#define NSLICES (NLAT / SLICE)     // 128
#define MBLOCKS (BATCH / MT)       // 32
#define CPR (NSLICES * TOPK)       // 8192 candidates per row
#define KC  64                     // K chunk staged in LDS (double buffered)

typedef __attribute__((ext_vector_type(16))) __bf16 v16bf;
typedef __attribute__((ext_vector_type(8)))  __bf16 v8bf;
typedef __attribute__((ext_vector_type(8)))  float  v8f;

// ---- CDNA5 async global->LDS path (guarded; falls back to plain copies) ----
#if defined(__has_builtin)
#if __has_builtin(__builtin_amdgcn_global_load_async_to_lds_b128) && \
    __has_builtin(__builtin_amdgcn_s_wait_asynccnt)
#define HAVE_ASYNC_LDS 1
#endif
#if __has_builtin(__builtin_amdgcn_tensor_load_to_lds) && \
    __has_builtin(__builtin_amdgcn_s_wait_tensorcnt)
#define HAVE_TDM 1
#endif
#endif
#ifndef HAVE_ASYNC_LDS
#define HAVE_ASYNC_LDS 0
#endif
#ifndef HAVE_TDM
#define HAVE_TDM 0
#endif

#if HAVE_ASYNC_LDS
// Exact parameter types per hipcc diagnostic (round 1):
//   param0: int __vector_size(16) addrspace(1)*, param1: addrspace(3)*
typedef int v4i_t __attribute__((vector_size(16)));
typedef __attribute__((address_space(1))) v4i_t g_v4i;
typedef __attribute__((address_space(3))) v4i_t l_v4i;
#endif

#if HAVE_TDM
typedef unsigned int v4u_tdm __attribute__((ext_vector_type(4)));
typedef int          v8i_tdm __attribute__((ext_vector_type(8)));
typedef int          v4i_tdm __attribute__((ext_vector_type(4)));
typedef __attribute__((address_space(3))) void lvoid_t;
#endif

// Monotone float->uint mapping so (mapped<<32|idx) u64 compares give float order
__device__ __forceinline__ unsigned fmap(float f) {
  unsigned u = __float_as_uint(f);
  return (u & 0x80000000u) ? ~u : (u | 0x80000000u);
}
__device__ __forceinline__ float funmap(unsigned m) {
  unsigned u = (m & 0x80000000u) ? (m ^ 0x80000000u) : ~m;
  return __uint_as_float(u);
}
__device__ __forceinline__ unsigned long long shflx64(unsigned long long v, int m) {
  int lo = __shfl_xor((int)(v & 0xffffffffull), m, 32);
  int hi = __shfl_xor((int)(v >> 32), m, 32);
  return ((unsigned long long)(unsigned)hi << 32) | (unsigned)lo;
}

// ---------------- prep: f32 -> bf16 conversions ----------------
__global__ __launch_bounds__(256)
void sae_prep_x(const float* __restrict__ x, const float* __restrict__ b_dec,
                __bf16* __restrict__ xb) {
  size_t i = ((size_t)blockIdx.x * 256 + threadIdx.x) * 8;
  int col = (int)(i & (DIN - 1));
  float4 v0 = *(const float4*)(x + i);
  float4 v1 = *(const float4*)(x + i + 4);
  float4 b0 = *(const float4*)(b_dec + col);
  float4 b1 = *(const float4*)(b_dec + col + 4);
  v8bf o;
  o[0] = (__bf16)(v0.x - b0.x); o[1] = (__bf16)(v0.y - b0.y);
  o[2] = (__bf16)(v0.z - b0.z); o[3] = (__bf16)(v0.w - b0.w);
  o[4] = (__bf16)(v1.x - b1.x); o[5] = (__bf16)(v1.y - b1.y);
  o[6] = (__bf16)(v1.z - b1.z); o[7] = (__bf16)(v1.w - b1.w);
  *(v8bf*)(xb + i) = o;
}

__global__ __launch_bounds__(256)
void sae_prep_w(const float* __restrict__ W, __bf16* __restrict__ wb) {
  size_t i = ((size_t)blockIdx.x * 256 + threadIdx.x) * 8;
  float4 v0 = *(const float4*)(W + i);
  float4 v1 = *(const float4*)(W + i + 4);
  v8bf o;
  o[0] = (__bf16)v0.x; o[1] = (__bf16)v0.y; o[2] = (__bf16)v0.z; o[3] = (__bf16)v0.w;
  o[4] = (__bf16)v1.x; o[5] = (__bf16)v1.y; o[6] = (__bf16)v1.z; o[7] = (__bf16)v1.w;
  *(v8bf*)(wb + i) = o;
}

// Stage one [MT x KC] bf16 x-tile into LDS buffer p (async when available)
__device__ __forceinline__ void stage_tile(const __bf16* __restrict__ xb,
                                           __bf16* As, int RB, int kc, int p, int tid) {
  int r  = tid >> 1;
  int ch = (tid & 1) * 32;
  const __bf16* src = xb + (size_t)(RB + r) * DIN + kc + ch;
  __bf16* dst = As + p * (MT * KC) + r * KC + ch;
#if HAVE_ASYNC_LDS
  #pragma unroll
  for (int q = 0; q < 4; ++q)
    __builtin_amdgcn_global_load_async_to_lds_b128(
        (g_v4i*)(src + q * 8), (l_v4i*)(dst + q * 8), 0, 0);
#else
  #pragma unroll
  for (int q = 0; q < 4; ++q)
    *(uint4*)(dst + q * 8) = *(const uint4*)(src + q * 8);
#endif
}

// ------- encode GEMM (bf16 WMMA) fused with per-slice top-64 -------
// grid = (NSLICES, MBLOCKS), block = 256 (8 waves, 2Mx4N wave grid)
// wave tile = 64 rows x 128 latents -> acc[4][8] v8f (256 VGPRs)
__global__ __launch_bounds__(256)
void sae_encode_topk_slice(const __bf16* __restrict__ xb,
                           const __bf16* __restrict__ wb,
                           const float* __restrict__ b_enc,
                           unsigned long long* __restrict__ cand) {
  __shared__ __align__(16) unsigned char smem[16 * SLICE * 4]; // 32KB, phase-shared
  __bf16* As  = (__bf16*)smem;   // 2 x [MT][KC] bf16 during GEMM (2 x 16KB)
  float*  buf = (float*)smem;    // [16][SLICE] f32 during selection (32KB)

  const int tid  = threadIdx.x;
  const int lane = tid & 31;
  const int w    = tid >> 5;      // wave 0..7
  const int wm   = w >> 2;        // 0..1 (M group)
  const int wn   = w & 3;         // 0..3 (N group)
  const int lr   = lane & 15;
  const int hig  = lane >> 4;     // 0: lanes0-15, 1: lanes16-31

  const int RB = blockIdx.y * MT;      // batch row base
  const int CB = blockIdx.x * SLICE;   // latent base (slice)

  v8f acc[4][8];
  #pragma unroll
  for (int mi = 0; mi < 4; ++mi)
    #pragma unroll
    for (int ni = 0; ni < 8; ++ni)
      acc[mi][ni] = 0.0f;

  // ---- K loop, double-buffered LDS-staged A tile ----
  stage_tile(xb, As, RB, 0, 0, tid);
  for (int kc = 0; kc < DIN; kc += KC) {
    const int p = (kc / KC) & 1;
    if (kc + KC < DIN) {
      stage_tile(xb, As, RB, kc + KC, p ^ 1, tid);
#if HAVE_ASYNC_LDS
      __builtin_amdgcn_s_wait_asynccnt(4);   // current chunk done, next in flight
#endif
    } else {
#if HAVE_ASYNC_LDS
      __builtin_amdgcn_s_wait_asynccnt(0);
#endif
    }
    __syncthreads();
    const __bf16* Ab = As + p * (MT * KC);

    #pragma unroll
    for (int kk = 0; kk < KC; kk += 32) {
      const int kstep = kc + kk;
      // A fragments (16x32 bf16): lane holds K[kq..kq+7] and K[kq+16..kq+23]
      v16bf a[4];
      #pragma unroll
      for (int mi = 0; mi < 4; ++mi) {
        const __bf16* ap = Ab + (wm * 64 + mi * 16 + lr) * KC + kk + hig * 8;
        v8bf a0 = *(const v8bf*)ap;
        v8bf a1 = *(const v8bf*)(ap + 16);
        a[mi] = __builtin_shufflevector(a0, a1, 0,1,2,3,4,5,6,7,8,9,10,11,12,13,14,15);
      }
      #pragma unroll
      for (int ni = 0; ni < 8; ++ni) {
        // B fragment (32x16 bf16): lane n holds K[kq..kq+7] and K[kq+16..kq+23] of W_enc row
        const __bf16* bp = wb + (size_t)(CB + wn * 128 + ni * 16 + lr) * DIN
                              + kstep + hig * 8;
        v8bf b0 = *(const v8bf*)bp;
        v8bf b1 = *(const v8bf*)(bp + 16);
        v16bf b = __builtin_shufflevector(b0, b1, 0,1,2,3,4,5,6,7,8,9,10,11,12,13,14,15);
        #pragma unroll
        for (int mi = 0; mi < 4; ++mi)
          acc[mi][ni] = __builtin_amdgcn_wmma_f32_16x16x32_bf16(
              false, a[mi], false, b, (short)0, acc[mi][ni], false, false);
      }
    }
    __syncthreads();
  }

  // ---- + b_enc (N depends only on lane) ----
  #pragma unroll
  for (int ni = 0; ni < 8; ++ni) {
    float be = b_enc[CB + wn * 128 + ni * 16 + lr];
    #pragma unroll
    for (int mi = 0; mi < 4; ++mi)
      #pragma unroll
      for (int c = 0; c < 8; ++c)
        acc[mi][ni][c] += be;
  }

  // ---- per-slice top-64: stage 16 rows at a time to LDS, wave argmax ----
  const float NEG_INF = __uint_as_float(0xff800000u);
  for (int g = 0; g < 8; ++g) {           // 8 groups of 16 rows
    __syncthreads();
    if ((g >> 2) == wm) {
      int mi = g & 3;
      #pragma unroll
      for (int ni = 0; ni < 8; ++ni) {
        int col = wn * 128 + ni * 16 + lr;
        #pragma unroll
        for (int c = 0; c < 8; ++c)
          buf[(c + hig * 8) * SLICE + col] = acc[mi][ni][c];
      }
    }
    __syncthreads();
    // wave w selects for rows 2w, 2w+1 of this group
    for (int rr = 2 * w; rr <= 2 * w + 1; ++rr) {
      float v[16];
      #pragma unroll
      for (int j = 0; j < 16; ++j) v[j] = buf[rr * SLICE + lane + 32 * j];
      unsigned long long lbest = 0;
      #pragma unroll
      for (int j = 0; j < 16; ++j) {
        unsigned long long key =
            ((unsigned long long)fmap(v[j]) << 32) | (unsigned)(j * 32 + lane);
        if (key > lbest) lbest = key;
      }
      size_t orow = (size_t)(RB + g * 16 + rr);
      unsigned long long* crow = cand + orow * (size_t)CPR + (size_t)blockIdx.x * TOPK;
      for (int t = 0; t < TOPK; ++t) {
        unsigned long long best = lbest;
        #pragma unroll
        for (int d = 1; d < 32; d <<= 1) {
          unsigned long long o = shflx64(best, d);
          if (o > best) best = o;
        }
        unsigned csel = (unsigned)best & 511u;   // local column in slice
        if ((csel & 31u) == (unsigned)lane) {    // owner removes + rescans
          v[csel >> 5] = NEG_INF;
          lbest = 0;
          #pragma unroll
          for (int j = 0; j < 16; ++j) {
            unsigned long long key =
                ((unsigned long long)fmap(v[j]) << 32) | (unsigned)(j * 32 + lane);
            if (key > lbest) lbest = key;
          }
        }
        if (lane == 0)
          crow[t] = (best & 0xffffffff00000000ull) | (unsigned)(CB + (int)csel);
      }
    }
  }
}

// ------- merge 128x64 slice candidates per row -> global top-64 (descending) -------
__global__ __launch_bounds__(256)
void sae_topk_merge(const unsigned long long* __restrict__ cand,
                    float* __restrict__ acts, int* __restrict__ idx) {
  const int r = blockIdx.x, t = threadIdx.x;
  const int lane = t & 31, w = t >> 5;
  const unsigned long long* base = cand + (size_t)r * CPR;
  unsigned long long keys[32];
  #pragma unroll
  for (int j = 0; j < 32; ++j) keys[j] = base[t + 256 * j];
  unsigned long long lmax = 0;
  #pragma unroll
  for (int j = 0; j < 32; ++j) if (keys[j] > lmax) lmax = keys[j];

  __shared__ unsigned long long red[8];
  for (int s = 0; s < TOPK; ++s) {
    unsigned long long m = lmax;
    #pragma unroll
    for (int d = 1; d < 32; d <<= 1) {
      unsigned long long o = shflx64(m, d);
      if (o > m) m = o;
    }
    if (lane == 0) red[w] = m;
    __syncthreads();
    unsigned long long g = red[0];
    #pragma unroll
    for (int i = 1; i < 8; ++i) if (red[i] > g) g = red[i];
    if (lmax == g) {                     // unique winner (keys are unique)
      #pragma unroll
      for (int j = 0; j < 32; ++j) if (keys[j] == g) keys[j] = 0;
      lmax = 0;
      #pragma unroll
      for (int j = 0; j < 32; ++j) if (keys[j] > lmax) lmax = keys[j];
      acts[r * TOPK + s] = funmap((unsigned)(g >> 32));
      idx[r * TOPK + s]  = (int)(g & 0xffffffffu);
    }
    __syncthreads();
  }
}

// ------- decode: out = sum_k acts*W_dec[idx] + b_dec -------
// TDM gather mode: one tensor_load_to_lds fetches 16 gathered W_dec rows x 512 cols
__global__ __launch_bounds__(256)
void sae_decode(const float* __restrict__ acts, const int* __restrict__ idx,
                const float* __restrict__ W_dec, const float* __restrict__ b_dec,
                float* __restrict__ out) {
  __shared__ float sA[TOPK];
  __shared__ int   sI[TOPK];
  const int r = blockIdx.x, t = threadIdx.x;
  if (t < TOPK) { sA[t] = acts[r * TOPK + t]; sI[t] = idx[r * TOPK + t]; }

#if HAVE_TDM
  __shared__ __align__(16) float sRows[16 * 512];       // 32KB gathered tile
  __shared__ unsigned short sI16[TOPK];
  if (t < TOPK) sI16[t] = (unsigned short)idx[r * TOPK + t];
  __syncthreads();

  for (int c = 0; c < 4; ++c) {                         // 512-column chunks
    const int col = c * 512 + 2 * t;                    // 2 columns per thread
    float a0 = b_dec[col], a1 = b_dec[col + 1];
    for (int g = 0; g < 4; ++g) {                       // 16 gathered rows per TDM op
      if (t < 32) {                                     // wave 0 issues the DMA
        unsigned long long ga =
            (unsigned long long)(size_t)W_dec + (unsigned long long)(c * 512) * 4ull;
        unsigned ldsoff = (unsigned)(unsigned long long)(lvoid_t*)&sRows[0];
        v4u_tdm g0;
        g0[0] = 0x80000001u;                            // count=1, gather, 16b idx
        g0[1] = ldsoff;                                 // lds_addr
        g0[2] = (unsigned)(ga & 0xffffffffu);           // global_addr[31:0]
        g0[3] = (unsigned)((ga >> 32) & 0x01ffffffu) | (2u << 30); // addr[56:32]|type=2
        v8i_tdm g1;
        g1[0] = 0x00020000;                             // data_size=2 (4B)
        g1[1] = (int)(2048u << 16);                     // tensor_dim0=2048 (lo16)
        g1[2] = 0;                                      // dim0 hi16, dim1 lo16
        g1[3] = (int)((512u << 16) | 1u);               // tile_dim0=512, dim1 hi16
        g1[4] = 16;                                     // tile_dim1 = #indices
        g1[5] = 2048;                                   // tensor_dim0_stride
        g1[6] = 0; g1[7] = 0;
        const unsigned short* ip = &sI16[g * 16];
        v4i_tdm g2, g3;
        #pragma unroll
        for (int q = 0; q < 4; ++q) {
          g2[q] = (int)((unsigned)ip[2*q]     | ((unsigned)ip[2*q + 1] << 16));
          g3[q] = (int)((unsigned)ip[8 + 2*q] | ((unsigned)ip[8 + 2*q + 1] << 16));
        }
        v8i_tdm g4 = {0, 0, 0, 0, 0, 0, 0, 0};          // 6-arg form: zero-filled
        __builtin_amdgcn_tensor_load_to_lds(g0, g1, g2, g3, g4, 0);
        __builtin_amdgcn_s_wait_tensorcnt(0);
      }
      __syncthreads();
      #pragma unroll 4
      for (int j = 0; j < 16; ++j) {
        float a = sA[g * 16 + j];
        a0 = fmaf(a, sRows[j * 512 + 2 * t], a0);
        a1 = fmaf(a, sRows[j * 512 + 2 * t + 1], a1);
      }
      __syncthreads();                                  // before sRows is rewritten
    }
    out[(size_t)r * DIN + col]     = a0;
    out[(size_t)r * DIN + col + 1] = a1;
  }
#else
  __syncthreads();
  const int c0 = t * 8;
  float4 a0 = *(const float4*)(b_dec + c0);
  float4 a1 = *(const float4*)(b_dec + c0 + 4);
  #pragma unroll 4
  for (int k = 0; k < TOPK; ++k) {
    if (k + 1 < TOPK)
      __builtin_prefetch(W_dec + (size_t)sI[k + 1] * DIN + c0, 0, 0);
    float a = sA[k];
    const float* wr = W_dec + (size_t)sI[k] * DIN + c0;
    float4 w0 = *(const float4*)wr;
    float4 w1 = *(const float4*)(wr + 4);
    a0.x = fmaf(a, w0.x, a0.x); a0.y = fmaf(a, w0.y, a0.y);
    a0.z = fmaf(a, w0.z, a0.z); a0.w = fmaf(a, w0.w, a0.w);
    a1.x = fmaf(a, w1.x, a1.x); a1.y = fmaf(a, w1.y, a1.y);
    a1.z = fmaf(a, w1.z, a1.z); a1.w = fmaf(a, w1.w, a1.w);
  }
  *(float4*)(out + (size_t)r * DIN + c0)     = a0;
  *(float4*)(out + (size_t)r * DIN + c0 + 4) = a1;
#endif
}

// ---------------- host launch ----------------
extern "C" void kernel_launch(void* const* d_in, const int* in_sizes, int n_in,
                              void* d_out, int out_size, void* d_ws, size_t ws_size,
                              hipStream_t stream) {
  const float* x     = (const float*)d_in[0];
  const float* W_enc = (const float*)d_in[1];
  const float* b_enc = (const float*)d_in[2];
  const float* W_dec = (const float*)d_in[3];
  const float* b_dec = (const float*)d_in[4];
  float* out = (float*)d_out;

  char* ws = (char*)d_ws;
  const size_t OFF_XB   = 0;                                  // 16 MB
  const size_t OFF_WB   = (size_t)BATCH * DIN * 2;            // +256 MB
  const size_t OFF_CAND = OFF_WB + (size_t)NLAT * DIN * 2;    // +256 MB
  const size_t OFF_ACT  = OFF_CAND + (size_t)BATCH * CPR * 8; // +1 MB
  const size_t OFF_IDX  = OFF_ACT + (size_t)BATCH * TOPK * 4; // +1 MB

  __bf16* xb   = (__bf16*)(ws + OFF_XB);
  __bf16* wbuf = (__bf16*)(ws + OFF_WB);
  unsigned long long* cand = (unsigned long long*)(ws + OFF_CAND);
  float* acts = (float*)(ws + OFF_ACT);
  int*   idx  = (int*)(ws + OFF_IDX);

  sae_prep_x<<<(BATCH * DIN) / 2048, 256, 0, stream>>>(x, b_dec, xb);
  sae_prep_w<<<((size_t)NLAT * DIN) / 2048, 256, 0, stream>>>(W_enc, wbuf);
  sae_encode_topk_slice<<<dim3(NSLICES, MBLOCKS), 256, 0, stream>>>(xb, wbuf, b_enc, cand);
  sae_topk_merge<<<BATCH, 256, 0, stream>>>(cand, acts, idx);
  sae_decode<<<BATCH, 256, 0, stream>>>(acts, idx, W_dec, b_dec, out);
}
